// ResRGATCell_31877247271041
// MI455X (gfx1250) — compile-verified, matrix operand order checked by reading, not run on hardware
//
#include <hip/hip_runtime.h>

#define N_NODES 20000
#define D_EDGES 16
#define HDIM 128
#define RDIM 128
#define INDIM 256
#define WAVES_PER_BLOCK 2
#define LDS_PER_WAVE 24576  // 16KB f32 [16][256] hsrel + 8KB bf16 [16][256] abf

typedef __attribute__((ext_vector_type(16))) __bf16 v16bf;
typedef __attribute__((ext_vector_type(8)))  float  v8f;

#if defined(__has_builtin)
#if __has_builtin(__builtin_amdgcn_tensor_load_to_lds) && __has_builtin(__builtin_amdgcn_s_wait_tensorcnt)
#define HAVE_TDM 1
#endif
#if __has_builtin(__builtin_amdgcn_cvt_pk_bf16_f32)
#define HAVE_CVTPK 1
#endif
#endif

// round-half-up bf16 truncate (differs from RNE only on exact ties)
__device__ __forceinline__ unsigned int f2bf(float f) {
  return (__float_as_uint(f) + 0x8000u) >> 16;
}

// pack two f32 -> two bf16 in one DWORD: 2x v_add + 1x v_perm_b32
__device__ __forceinline__ unsigned int packbf2(float a, float b) {
#if defined(HAVE_CVTPK)
  auto r = __builtin_amdgcn_cvt_pk_bf16_f32(a, b);
  unsigned int u;
  __builtin_memcpy(&u, &r, sizeof(u));
  return u;
#else
  unsigned int au = __float_as_uint(a) + 0x8000u;
  unsigned int bu = __float_as_uint(b) + 0x8000u;
  // v_perm_b32: S1 supplies bytes 0-3, S0 supplies bytes 4-7; take hi16 of each
  return __builtin_amdgcn_perm(bu, au, 0x07060302u);
#endif
}

union FragU {
  uint4 q[2];
  unsigned int u[8];
  v16bf v;
};

// A-fragment: 16x32 bf16, row-major source in LDS (ldk bf16 per row).
// Lane L holds row M=L%16; K offsets {0..7,16..23} (lanes 0-15) or {8..15,24..31} (lanes 16-31).
__device__ __forceinline__ v16bf load_afrag(const unsigned short* base, int ldk, int kb, int lane) {
  int m = lane & 15;
  int koff = kb * 32 + ((lane & 16) ? 8 : 0);
  const unsigned short* p = base + m * ldk + koff;
  FragU u;
  u.q[0] = *(const uint4*)(p);
  u.q[1] = *(const uint4*)(p + 16);
  return u.v;
}

// B-fragment from prepped global layout: frag id = nt*nkb+kb; 32 lanes x 16 bf16 contiguous.
__device__ __forceinline__ v16bf load_bfrag(const unsigned short* frags, int nt, int kb, int nkb, int lane) {
  const unsigned short* p = frags + (((size_t)(nt * nkb + kb) * 32 + lane) * 16);
  FragU u;
  u.q[0] = *(const uint4*)(p);
  u.q[1] = *(const uint4*)(p + 8);
  return u.v;
}

__device__ __forceinline__ v8f bwmma(v16bf a, v16bf b, v8f c) {
  return __builtin_amdgcn_wmma_f32_16x16x32_bf16(false, a, false, b, (short)0, c, false, false);
}

#if defined(HAVE_TDM)
// TDM gather-mode load: 16 indexed rows of `dim0` f32 from a [dim1 x dim0] tensor into LDS.
// pad_enable inserts 128 DWORDs (512B) after every 128 DWORDs (512B) stored -> 1024B LDS row stride.
__device__ __forceinline__ void tdm_gather16(const void* gbase, unsigned int lds_byte,
                                             unsigned int dim0, unsigned int dim1,
                                             const unsigned int* w /*8 packed u16 index pairs*/) {
  typedef unsigned int u32x4 __attribute__((ext_vector_type(4)));
  typedef int i32x8 __attribute__((ext_vector_type(8)));
  typedef int i32x4 __attribute__((ext_vector_type(4)));
  unsigned long long ga = (unsigned long long)gbase;
  u32x4 g0;
  g0[0] = 0x80000001u;  // gather_mode=1, 16-bit indices, count=1
  g0[1] = lds_byte;     // lds_addr
  g0[2] = (unsigned int)(ga & 0xFFFFFFFFu);
  g0[3] = ((unsigned int)(ga >> 32) & 0x01FFFFFFu) | (2u << 30);  // global_addr hi, type=2
  i32x8 g1;
  // data_size=2 (4B), pad_enable, pad_interval=6 (128 DW), pad_amount=127 (128 DW)
  g1[0] = (int)((2u << 16) | (1u << 20) | (6u << 22) | (127u << 25));
  g1[1] = (int)((dim0 & 0xFFFFu) << 16);                                   // tensor_dim0 lo
  g1[2] = (int)(((dim0 >> 16) & 0xFFFFu) | ((dim1 & 0xFFFFu) << 16));      // dim0 hi | tensor_dim1 lo
  g1[3] = (int)(((dim1 >> 16) & 0xFFFFu) | ((dim0 & 0xFFFFu) << 16));      // dim1 hi | tile_dim0
  g1[4] = 16;                                                              // tile_dim1 = #indices
  g1[5] = (int)dim0;                                                       // tensor_dim0_stride lo32
  g1[6] = 0;
  g1[7] = 0;
  i32x4 g2, g3;
  g2[0] = (int)w[0]; g2[1] = (int)w[1]; g2[2] = (int)w[2]; g2[3] = (int)w[3];
  g3[0] = (int)w[4]; g3[1] = (int)w[5]; g3[2] = (int)w[6]; g3[3] = (int)w[7];
  i32x8 g4 = {};  // unused extra group (6-arg clang-23 form)
  __builtin_amdgcn_tensor_load_to_lds(g0, g1, g2, g3, g4, 0);
}
#endif

// ---------------- weight prep: f32 [K x 128] row-major -> bf16 B-fragment layout ----------------
__global__ void prep_frags(const float* __restrict__ W, unsigned short* __restrict__ dst, int K) {
  int nkb = K >> 5;
  int total = 8 * nkb * 32 * 16;
  int i = blockIdx.x * blockDim.x + threadIdx.x;
  if (i >= total) return;
  int e = i & 15;
  int lane = (i >> 4) & 31;
  int fid = i >> 9;
  int kb = fid % nkb;
  int nt = fid / nkb;
  int n = nt * 16 + (lane & 15);
  int koff = kb * 32 + ((lane & 16) ? 8 : 0);
  int k = koff + e + ((e >= 8) ? 8 : 0);  // e 0..7 -> +0..7 ; e 8..15 -> +16..23
  dst[i] = (unsigned short)f2bf(W[(size_t)k * 128 + n]);
}

// ---------------- Q = h @ Wq  (M=20000, K=128, N=128) ----------------
__global__ __launch_bounds__(256) void qgemm(const float* __restrict__ h,
                                             const unsigned short* __restrict__ wqf,
                                             float* __restrict__ Q) {
  int lane = threadIdx.x & 31;
  int ntile = threadIdx.x >> 5;  // 8 waves -> 8 N-tiles
  int mtile = blockIdx.x;
  int m = mtile * 16 + (lane & 15);
  int off = (lane & 16) ? 8 : 0;
  v8f acc = {};
#pragma unroll
  for (int kb = 0; kb < 4; ++kb) {
    int koff = kb * 32 + off;
    const float* pa = h + (size_t)m * 128 + koff;
    float4 a0 = *(const float4*)(pa);
    float4 a1 = *(const float4*)(pa + 4);
    float4 a2 = *(const float4*)(pa + 16);
    float4 a3 = *(const float4*)(pa + 20);
    FragU A;
    A.u[0] = packbf2(a0.x, a0.y); A.u[1] = packbf2(a0.z, a0.w);
    A.u[2] = packbf2(a1.x, a1.y); A.u[3] = packbf2(a1.z, a1.w);
    A.u[4] = packbf2(a2.x, a2.y); A.u[5] = packbf2(a2.z, a2.w);
    A.u[6] = packbf2(a3.x, a3.y); A.u[7] = packbf2(a3.z, a3.w);
    v16bf B = load_bfrag(wqf, ntile, kb, 4, lane);
    acc = bwmma(A.v, B, acc);
  }
  int col = ntile * 16 + (lane & 15);
#pragma unroll
  for (int v = 0; v < 8; ++v) {
    int row = mtile * 16 + v + off;
    Q[(size_t)row * 128 + col] = acc[v];
  }
}

// ---------------- main per-node kernel: one wave per node ----------------
__global__ __launch_bounds__(WAVES_PER_BLOCK * 32) void node_kernel(
    const float* __restrict__ h, const int* __restrict__ src_idx, const int* __restrict__ rel_id,
    const float* __restrict__ relvec, const float* __restrict__ ln_g, const float* __restrict__ ln_b,
    const unsigned short* __restrict__ waf, const float* __restrict__ bA,
    const unsigned short* __restrict__ wbf, const float* __restrict__ bB,
    const unsigned short* __restrict__ wkf, const float* __restrict__ Wq,
    const float* __restrict__ Qbuf, float* __restrict__ out) {
  extern __shared__ char lds[];
  int lane = threadIdx.x & 31;
  int wave = threadIdx.x >> 5;
  int node = blockIdx.x * WAVES_PER_BLOCK + wave;

  float* hsrel = (float*)(lds + wave * LDS_PER_WAVE);                  // [16][256] f32: [hs | relv]
  unsigned short* abf = (unsigned short*)(lds + wave * LDS_PER_WAVE + 16384);  // [16][256] bf16

  const int half = (lane >> 4);
  const int l16 = lane & 15;
  const int off = half ? 8 : 0;

  // ---- 1. gather hs + relv into LDS ----
#if defined(HAVE_TDM)
  {
    unsigned int ws_[8], wr_[8];
#pragma unroll
    for (int j = 0; j < 8; ++j) {
      unsigned int s0 = (unsigned int)src_idx[node * 16 + 2 * j] & 0xFFFFu;
      unsigned int s1 = (unsigned int)src_idx[node * 16 + 2 * j + 1] & 0xFFFFu;
      unsigned int r0 = (unsigned int)rel_id[node * 16 + 2 * j] & 0xFFFFu;
      unsigned int r1 = (unsigned int)rel_id[node * 16 + 2 * j + 1] & 0xFFFFu;
      ws_[j] = (unsigned int)__builtin_amdgcn_readfirstlane((int)(s0 | (s1 << 16)));
      wr_[j] = (unsigned int)__builtin_amdgcn_readfirstlane((int)(r0 | (r1 << 16)));
    }
    unsigned int base = (unsigned int)__builtin_amdgcn_readfirstlane(wave * LDS_PER_WAVE);
    tdm_gather16(h, base, 128u, (unsigned int)N_NODES, ws_);        // hs rows -> cols 0..127
    tdm_gather16(relvec, base + 512u, 128u, 32u, wr_);              // relv rows -> cols 128..255
    __builtin_amdgcn_s_wait_tensorcnt(0);
  }
#else
  for (int r = 0; r < 16; ++r) {
    int s = src_idx[node * 16 + r];
    int rid = rel_id[node * 16 + r];
    float4 hv = *(const float4*)(h + (size_t)s * 128 + lane * 4);
    float4 rv = *(const float4*)(relvec + (size_t)rid * 128 + lane * 4);
    *(float4*)(hsrel + r * 256 + lane * 4) = hv;
    *(float4*)(hsrel + r * 256 + 128 + lane * 4) = rv;
  }
#endif
  __syncthreads();

  // ---- 2. LayerNorm over 256, write bf16 A operand to abf[16][256] ----
  {
    const float* row = hsrel + l16 * 256 + half * 128;
    float s = 0.f, sq = 0.f;
    for (int c = 0; c < 128; c += 4) {
      float4 x = *(const float4*)(row + c);
      s += x.x + x.y + x.z + x.w;
      sq += x.x * x.x + x.y * x.y + x.z * x.z + x.w * x.w;
    }
    s += __shfl_xor(s, 16);
    sq += __shfl_xor(sq, 16);
    float mean = s * (1.f / 256.f);
    float var = sq * (1.f / 256.f) - mean * mean;
    float inv = rsqrtf(var + 1e-5f);
    for (int c = 0; c < 128; c += 4) {
      int col = half * 128 + c;
      float4 x = *(const float4*)(row + c);
      float4 g = *(const float4*)(ln_g + col);
      float4 b = *(const float4*)(ln_b + col);
      uint2 pk;
      pk.x = packbf2((x.x - mean) * inv * g.x + b.x, (x.y - mean) * inv * g.y + b.y);
      pk.y = packbf2((x.z - mean) * inv * g.z + b.z, (x.w - mean) * inv * g.w + b.w);
      *(uint2*)&abf[l16 * 256 + col] = pk;
    }
  }
  __syncthreads();

  // ---- 3. GEMM A: [16x256]*[256x128] + bias, CELU -> abf as [16][128] bf16 ----
  v8f acc[8];
#pragma unroll
  for (int nt = 0; nt < 8; ++nt) acc[nt] = (v8f){};
#pragma unroll
  for (int kb = 0; kb < 8; ++kb) {
    v16bf A = load_afrag(abf, 256, kb, lane);
#pragma unroll
    for (int nt = 0; nt < 8; ++nt) {
      v16bf B = load_bfrag(waf, nt, kb, 8, lane);
      acc[nt] = bwmma(A, B, acc[nt]);
    }
  }
  __syncthreads();
#pragma unroll
  for (int nt = 0; nt < 8; ++nt) {
    int col = nt * 16 + l16;
    float bias = bA[col];
#pragma unroll
    for (int v = 0; v < 8; ++v) {
      int row = v + off;
      float x = acc[nt][v] + bias;
      float cel = x > 0.f ? x : expm1f(x);  // CELU alpha=1
      abf[row * 128 + col] = (unsigned short)f2bf(cel);
    }
  }
  __syncthreads();

  // ---- 4. GEMM B: [16x128]*[128x128] + bias + residual -> hs_new into hsrel[:, :128] ----
  v8f acc2[8];
#pragma unroll
  for (int nt = 0; nt < 8; ++nt) acc2[nt] = (v8f){};
#pragma unroll
  for (int kb = 0; kb < 4; ++kb) {
    v16bf A = load_afrag(abf, 128, kb, lane);
#pragma unroll
    for (int nt = 0; nt < 8; ++nt) {
      v16bf B = load_bfrag(wbf, nt, kb, 4, lane);
      acc2[nt] = bwmma(A, B, acc2[nt]);
    }
  }
  __syncthreads();
#pragma unroll
  for (int nt = 0; nt < 8; ++nt) {
    int col = nt * 16 + l16;
    float bias = bB[col];
#pragma unroll
    for (int v = 0; v < 8; ++v) {
      int row = v + off;
      float hn = hsrel[row * 256 + col] + acc2[nt][v] + bias;
      hsrel[row * 256 + col] = hn;  // hs_new (keys = [hs_new | relv] now lives in hsrel)
    }
  }
  __syncthreads();

  // ---- 5. keys -> bf16 operand in abf[16][256] (float4 loads, packed b64 stores) ----
  for (int i = 0; i < 32; ++i) {
    int j = i * 128 + lane * 4;
    float4 x = *(const float4*)(hsrel + j);
    uint2 pk;
    pk.x = packbf2(x.x, x.y);
    pk.y = packbf2(x.z, x.w);
    *(uint2*)&abf[j] = pk;
  }
  __syncthreads();

  // ---- 6. GEMM K: keys[16x256]*Wk[256x128] -> k ----
  v8f acck[8];
#pragma unroll
  for (int nt = 0; nt < 8; ++nt) acck[nt] = (v8f){};
#pragma unroll
  for (int kb = 0; kb < 8; ++kb) {
    v16bf A = load_afrag(abf, 256, kb, lane);
#pragma unroll
    for (int nt = 0; nt < 8; ++nt) {
      v16bf B = load_bfrag(wkf, nt, kb, 8, lane);
      acck[nt] = bwmma(A, B, acck[nt]);
    }
  }
  __syncthreads();
  float* kbuf = (float*)abf;  // reuse as f32 [16][128]
#pragma unroll
  for (int nt = 0; nt < 8; ++nt) {
    int col = nt * 16 + l16;
#pragma unroll
    for (int v = 0; v < 8; ++v) kbuf[(v + off) * 128 + col] = acck[nt][v];
  }

  // ---- 7. q into LDS (precomputed Q, or VALU fallback) ----
  if (Qbuf) {
#pragma unroll
    for (int cc = 0; cc < 4; ++cc) {
      int c = cc * 32 + lane;
      hsrel[384 + c] = Qbuf[(size_t)node * 128 + c];
    }
  } else {
#pragma unroll
    for (int cc = 0; cc < 4; ++cc) {
      int c = cc * 32 + lane;
      float s = 0.f;
      for (int j = 0; j < 128; ++j) s += h[(size_t)node * 128 + j] * Wq[(size_t)j * 128 + c];
      hsrel[384 + c] = s;
    }
  }
  __syncthreads();
  const float* qs = hsrel + 384;

  // ---- 8. attention: lane handles (h=half, s=l16) and (h=half+2, s=l16) ----
  {
    const float scale = 0.17677669529663687f;  // 1/sqrt(32)
    const float* krow = kbuf + l16 * 128;
    int h0 = half, h1 = half + 2;
    float w0 = 0.f, w1 = 0.f;
    for (int d = 0; d < 32; d += 4) {
      float4 q0 = *(const float4*)(qs + h0 * 32 + d);
      float4 k0 = *(const float4*)(krow + h0 * 32 + d);
      float4 q1 = *(const float4*)(qs + h1 * 32 + d);
      float4 k1 = *(const float4*)(krow + h1 * 32 + d);
      w0 += q0.x * k0.x + q0.y * k0.y + q0.z * k0.z + q0.w * k0.w;
      w1 += q1.x * k1.x + q1.y * k1.y + q1.z * k1.z + q1.w * k1.w;
    }
    w0 *= scale; w1 *= scale;
    float m0 = w0, m1 = w1;
    for (int msk = 8; msk >= 1; msk >>= 1) {
      m0 = fmaxf(m0, __shfl_xor(m0, msk));
      m1 = fmaxf(m1, __shfl_xor(m1, msk));
    }
    float e0 = __expf(w0 - m0), e1 = __expf(w1 - m1);
    float s0 = e0, s1 = e1;
    for (int msk = 8; msk >= 1; msk >>= 1) {
      s0 += __shfl_xor(s0, msk);
      s1 += __shfl_xor(s1, msk);
    }
    hsrel[128 + h0 * 16 + l16] = e0 / s0;  // a[h][s] in free relv slot of row 0
    hsrel[128 + h1 * 16 + l16] = e1 / s1;
  }
  __syncthreads();

  // ---- 9. red = a @ v (v = hs_new); lane -> d = lane, all 4 heads ----
  float o0 = 0.f, o1 = 0.f, o2 = 0.f, o3 = 0.f;
  for (int s = 0; s < 16; ++s) {
    const float* vrow = hsrel + s * 256;
    float4 a = *(const float4*)(hsrel + 128 + s * 0);  // placeholder avoided below
    (void)a;
    float a0 = hsrel[128 + 0 * 16 + s];
    float a1 = hsrel[128 + 1 * 16 + s];
    float a2 = hsrel[128 + 2 * 16 + s];
    float a3 = hsrel[128 + 3 * 16 + s];
    o0 += a0 * vrow[0 * 32 + lane];
    o1 += a1 * vrow[1 * 32 + lane];
    o2 += a2 * vrow[2 * 32 + lane];
    o3 += a3 * vrow[3 * 32 + lane];
  }
  out[(size_t)node * 128 + 0 * 32 + lane] = o0;
  out[(size_t)node * 128 + 1 * 32 + lane] = o1;
  out[(size_t)node * 128 + 2 * 32 + lane] = o2;
  out[(size_t)node * 128 + 3 * 32 + lane] = o3;
}

extern "C" void kernel_launch(void* const* d_in, const int* in_sizes, int n_in,
                              void* d_out, int out_size, void* d_ws, size_t ws_size,
                              hipStream_t stream) {
  const float* h       = (const float*)d_in[0];
  const int*   src_idx = (const int*)d_in[1];
  const int*   rel_id  = (const int*)d_in[2];
  const float* relvec  = (const float*)d_in[3];
  const float* ln_g    = (const float*)d_in[4];
  const float* ln_b    = (const float*)d_in[5];
  const float* WA      = (const float*)d_in[6];
  const float* bA      = (const float*)d_in[7];
  const float* WB      = (const float*)d_in[8];
  const float* bB      = (const float*)d_in[9];
  const float* Wq      = (const float*)d_in[10];
  const float* Wk      = (const float*)d_in[11];
  float* out = (float*)d_out;

  char* ws = (char*)d_ws;
  unsigned short* waf = (unsigned short*)(ws + 0);       // 8nt*8kb*512 bf16 = 64KB
  unsigned short* wbf = (unsigned short*)(ws + 65536);   // 8nt*4kb*512 bf16 = 32KB
  unsigned short* wkf = (unsigned short*)(ws + 98304);   // 64KB
  unsigned short* wqf = (unsigned short*)(ws + 163840);  // 32KB
  const size_t frag_bytes = 196608;
  const size_t q_bytes = (size_t)N_NODES * HDIM * sizeof(float);
  float* Qbuf = (ws_size >= frag_bytes + q_bytes) ? (float*)(ws + frag_bytes) : nullptr;

  prep_frags<<<(8 * 8 * 512 + 255) / 256, 256, 0, stream>>>(WA, waf, 256);
  prep_frags<<<(8 * 4 * 512 + 255) / 256, 256, 0, stream>>>(WB, wbf, 128);
  prep_frags<<<(8 * 8 * 512 + 255) / 256, 256, 0, stream>>>(Wk, wkf, 256);
  if (Qbuf) {
    prep_frags<<<(8 * 4 * 512 + 255) / 256, 256, 0, stream>>>(Wq, wqf, 128);
    qgemm<<<N_NODES / 16, 256, 0, stream>>>(h, wqf, Qbuf);
  }
  node_kernel<<<N_NODES / WAVES_PER_BLOCK, WAVES_PER_BLOCK * 32,
                WAVES_PER_BLOCK * LDS_PER_WAVE, stream>>>(
      h, src_idx, rel_id, relvec, ln_g, ln_b, waf, bA, wbf, bB, wkf, Wq, Qbuf, out);
}
